// MultiHeadAttention_34909494182365
// MI455X (gfx1250) — matrix operand order repaired and checked by reading, not compiled, not run on hardware
//
#include <hip/hip_runtime.h>
#include <hip/hip_bf16.h>
#include <hip/hip_fp16.h>

// MI455X / gfx1250, wave32. All matrix math via v_wmma_f32_16x16x32_f16.
// Attention stages K/V tiles into LDS (async global->LDS when available),
// shared by all 4 waves of a block (same b,h) -> 4x less K/V L2 traffic.
// Workspace requirement: 48 MiB.

typedef __attribute__((ext_vector_type(16))) _Float16 v16h;
typedef __attribute__((ext_vector_type(8)))  _Float16 v8h;
typedef __attribute__((ext_vector_type(8)))  float    v8f;
typedef __attribute__((ext_vector_type(4)))  float    v4f;
typedef __attribute__((ext_vector_type(4)))  int      v4i;

#define DIM    1024
#define HEADS  16
#define HDIM   64
#define SEQ    2048
#define BATCH  2
#define TOK    (BATCH * SEQ)      // 4096
#define NCOL3  (3 * DIM)          // 3072

// ---- async global -> LDS copy (CDNA5 ASYNCcnt path), with fallback ----------
// Builtin signature (probed via diagnostics): (global int4*, shared int4*, Ii, Ii)
typedef __attribute__((address_space(1))) v4i gbl_v4i;
typedef __attribute__((address_space(3))) v4i lds_v4i;

#if __has_builtin(__builtin_amdgcn_global_load_async_to_lds_b128)
#define HAVE_ASYNC_LDS 1
#define ASYNC_COPY_B128(g, l)                                                  \
  __builtin_amdgcn_global_load_async_to_lds_b128(                              \
      (gbl_v4i*)(g), (lds_v4i*)(l), 0, 0)
#else
#define HAVE_ASYNC_LDS 0
#define ASYNC_COPY_B128(g, l) (*(v8h*)(l) = *(const v8h*)(g))
#endif

__device__ __forceinline__ void wait_async_copies() {
#if HAVE_ASYNC_LDS
#if __has_builtin(__builtin_amdgcn_s_wait_asynccnt)
  __builtin_amdgcn_s_wait_asynccnt(0);
#else
  asm volatile("s_wait_asynccnt 0x0" ::: "memory");
#endif
#endif
}

// ---- CDNA5 WMMA fragment loaders (wave32) -----------------------------------
// A-matrix 16x32 f16: lane(0..15)=row M, hi=lane>>4.
//   halves 0..7  -> K = hi*8 + 0..7 ; halves 8..15 -> K = 16 + hi*8 + 0..7
__device__ __forceinline__ v16h load_a_frag(const _Float16* base, int ld) {
  const int lane = threadIdx.x & 31;
  const int m    = lane & 15;
  const int hi   = lane >> 4;
  const _Float16* p = base + (size_t)m * ld + hi * 8;
  v8h lo = *(const v8h*)(p);
  v8h hh = *(const v8h*)(p + 16);
  v16h r;
#pragma unroll
  for (int i = 0; i < 8; ++i) { r[i] = lo[i]; r[i + 8] = hh[i]; }
  return r;
}

// B-matrix 32x16 f16 from K-contiguous ("transposed") layout:
// element(k,n) = baseT[n*ld + k]; lane n = column; halves 0..7 -> K = hi*8 + 0..7 ; halves 8..15 -> K = 16 + hi*8 + 0..7
__device__ __forceinline__ v16h load_b_fragT(const _Float16* baseT, int ld) {
  const int lane = threadIdx.x & 31;
  const int n    = lane & 15;
  const int hi   = lane >> 4;
  const _Float16* p = baseT + (size_t)n * ld + hi * 8;
  v8h lo = *(const v8h*)(p);
  v8h hh = *(const v8h*)(p + 16);
  v16h r;
#pragma unroll
  for (int i = 0; i < 8; ++i) { r[i] = lo[i]; r[i + 8] = hh[i]; }
  return r;
}

__device__ __forceinline__ v8f wmma_f16(v16h a, v16h b, v8f c) {
  return __builtin_amdgcn_wmma_f32_16x16x32_f16(false, a, false, b, (short)0, c,
                                                false, false);
}

// ---- Kernel 1: LayerNorm (fp32 stats) -> f16 --------------------------------
__global__ void k_ln_f16(const float* __restrict__ x,
                         const float* __restrict__ gamma,
                         const float* __restrict__ beta,
                         _Float16* __restrict__ h16) {
  const int wave = blockIdx.x * (blockDim.x >> 5) + (threadIdx.x >> 5);
  const int lane = threadIdx.x & 31;
  const float* row = x + (size_t)wave * DIM + lane * 32;
  float vals[32];
  float s = 0.f, ss = 0.f;
#pragma unroll
  for (int i = 0; i < 8; ++i) {
    v4f v = ((const v4f*)row)[i];
#pragma unroll
    for (int j = 0; j < 4; ++j) {
      float f = v[j];
      vals[i * 4 + j] = f;
      s += f; ss += f * f;
    }
  }
#pragma unroll
  for (int off = 16; off >= 1; off >>= 1) {
    s  += __shfl_xor(s,  off, 32);
    ss += __shfl_xor(ss, off, 32);
  }
  const float mu   = s * (1.0f / DIM);
  const float var  = ss * (1.0f / DIM) - mu * mu;
  const float rstd = rsqrtf(var + 1e-5f);
  const float* g = gamma + lane * 32;
  const float* b = beta  + lane * 32;
  _Float16* orow = h16 + (size_t)wave * DIM + lane * 32;
#pragma unroll
  for (int v8 = 0; v8 < 4; ++v8) {
    v8h o;
#pragma unroll
    for (int j = 0; j < 8; ++j) {
      int i = v8 * 8 + j;
      o[j] = (_Float16)((vals[i] - mu) * rstd * g[i] + b[i]);
    }
    ((v8h*)orow)[v8] = o;
  }
}

// ---- Kernel 2: weight transpose + f32->f16:  wt[c][k] = (f16)w[k][c] --------
__global__ void k_wtrans(const float* __restrict__ w, _Float16* __restrict__ wt,
                         int K, int N) {
  size_t idx = (size_t)blockIdx.x * blockDim.x + threadIdx.x;
  size_t total = (size_t)K * N;
  if (idx < total) {
    int c = (int)(idx / K);
    int k = (int)(idx % K);
    wt[idx] = (_Float16)w[(size_t)k * N + c];
  }
}

// ---- Kernel 3: QKV GEMM  h16[4096x1024] @ wqkvT -> scatter Q/K/Vt -----------
__global__ void k_qkv_gemm(const _Float16* __restrict__ h16,
                           const _Float16* __restrict__ wqkvT,
                           _Float16* __restrict__ q,
                           _Float16* __restrict__ kmat,
                           _Float16* __restrict__ vt) {
  const int wid   = blockIdx.x * (blockDim.x >> 5) + (threadIdx.x >> 5);
  const int row0  = (wid / 48) * 16;
  const int col0  = (wid % 48) * 64;
  v8f acc[4] = {v8f{}, v8f{}, v8f{}, v8f{}};
  for (int kk = 0; kk < DIM; kk += 32) {
    __builtin_prefetch(h16 + (size_t)row0 * DIM + kk + 64, 0, 1);
    v16h a = load_a_frag(h16 + (size_t)row0 * DIM + kk, DIM);
#pragma unroll
    for (int t = 0; t < 4; ++t) {
      v16h b = load_b_fragT(wqkvT + (size_t)(col0 + t * 16) * DIM + kk, DIM);
      acc[t] = wmma_f16(a, b, acc[t]);
    }
  }
  const int lane = threadIdx.x & 31;
  const int n    = lane & 15;
  const int hi   = lane >> 4;
#pragma unroll
  for (int t = 0; t < 4; ++t) {
#pragma unroll
    for (int r = 0; r < 8; ++r) {
      const int col = col0 + t * 16 + n;
      const int row = row0 + r + 8 * hi;         // token index
      const int s  = col >> 10;                  // 0=q 1=k 2=v
      const int h  = (col & 1023) >> 6;
      const int d  = col & 63;
      const int b  = row >> 11;                  // batch
      const int nn = row & 2047;                 // position
      const size_t head = (size_t)(b * HEADS + h);
      const _Float16 val = (_Float16)acc[t][r];
      if (s == 0)      q   [head * SEQ * HDIM + (size_t)nn * HDIM + d] = val;
      else if (s == 1) kmat[head * SEQ * HDIM + (size_t)nn * HDIM + d] = val;
      else             vt  [head * HDIM * SEQ + (size_t)d * SEQ + nn]  = val;
    }
  }
}

// ---- Kernel 4: flash attention ----------------------------------------------
// Block = 4 waves, all same (b,h): K/V tiles staged once per block into LDS.
__global__ void k_attn(const _Float16* __restrict__ q,
                       const _Float16* __restrict__ kmat,
                       const _Float16* __restrict__ vt,
                       _Float16* __restrict__ attn16) {
  __shared__ _Float16 Ktile[32 * HDIM];          // [key-in-tile][d]   4 KB
  __shared__ _Float16 Vtile[HDIM * 32];          // [d][key-in-tile]   4 KB
  __shared__ _Float16 pstage[4][16 * 32];        // per-wave P staging 4 KB

  const int tid  = threadIdx.x;                  // 0..127
  const int wid  = blockIdx.x * (blockDim.x >> 5) + (tid >> 5);
  const int qt   = wid & 127;                    // query tile within head
  const int bh   = wid >> 7;                     // 0..31 (uniform in block)
  const int lane = tid & 31;
  const int n    = lane & 15;
  const int hi   = lane >> 4;
  _Float16* ps = pstage[tid >> 5];

  const _Float16* Q  = q    + (size_t)bh * SEQ * HDIM + (size_t)qt * 16 * HDIM;
  const _Float16* K  = kmat + (size_t)bh * SEQ * HDIM;
  const _Float16* Vt = vt   + (size_t)bh * HDIM * SEQ;

  const v16h qa0 = load_a_frag(Q + 0,  HDIM);    // d = 0..31
  const v16h qa1 = load_a_frag(Q + 32, HDIM);    // d = 32..63

  float mrow[8], lrow[8];
  v8f o[4] = {v8f{}, v8f{}, v8f{}, v8f{}};
#pragma unroll
  for (int r = 0; r < 8; ++r) { mrow[r] = -3.0e38f; lrow[r] = 0.f; }
  const float scale = 0.125f;                    // 1/sqrt(64)

  for (int k0 = 0; k0 < SEQ; k0 += 32) {
    // ---- cooperative K/V tile stage (async global->LDS) ----
    {
      const _Float16* gk = K + (size_t)k0 * HDIM;   // contiguous 4 KB
#pragma unroll
      for (int i = 0; i < 2; ++i) {
        const int off = (tid + i * 128) * 8;        // halfs, b128 chunks
        ASYNC_COPY_B128(gk + off, Ktile + off);
      }
#pragma unroll
      for (int i = 0; i < 2; ++i) {
        const int j = tid + i * 128;                // 0..255
        const int d = j >> 2, part = j & 3;
        ASYNC_COPY_B128(Vt + (size_t)d * SEQ + k0 + part * 8,
                        Vtile + d * 32 + part * 8);
      }
      __builtin_prefetch(K + (size_t)(k0 + 32) * HDIM, 0, 1);
      wait_async_copies();
      __syncthreads();
    }

    // ---- scores: 16 queries x 32 keys, fp32 accum (B-frags from LDS) ----
    v8f s0 = v8f{}, s1 = v8f{};
    s0 = wmma_f16(qa0, load_b_fragT(Ktile + 0,            HDIM), s0);
    s0 = wmma_f16(qa1, load_b_fragT(Ktile + 32,           HDIM), s0);
    s1 = wmma_f16(qa0, load_b_fragT(Ktile + 16 * HDIM,    HDIM), s1);
    s1 = wmma_f16(qa1, load_b_fragT(Ktile + 16 * HDIM + 32, HDIM), s1);

    // ---- online softmax; C/D layout: row M = r + 8*hi, col = n ----
#pragma unroll
    for (int r = 0; r < 8; ++r) {
      float v = fmaxf(s0[r], s1[r]) * scale;
#pragma unroll
      for (int off = 8; off >= 1; off >>= 1) v = fmaxf(v, __shfl_xor(v, off, 32));
      const float newm = fmaxf(mrow[r], v);
      const float corr = __expf(mrow[r] - newm);
      lrow[r] *= corr;
#pragma unroll
      for (int t = 0; t < 4; ++t) o[t][r] *= corr;
      const float p0 = __expf(s0[r] * scale - newm);
      const float p1 = __expf(s1[r] * scale - newm);
      s0[r] = p0; s1[r] = p1;
      float psum = p0 + p1;
#pragma unroll
      for (int off = 8; off >= 1; off >>= 1) psum += __shfl_xor(psum, off, 32);
      lrow[r] += psum;
      mrow[r] = newm;
    }

    // ---- re-layout P (C/D) -> row-major 16x32 in LDS, A-fragment load ----
#pragma unroll
    for (int r = 0; r < 8; ++r) {
      ps[(r + 8 * hi) * 32 + n]      = (_Float16)s0[r];
      ps[(r + 8 * hi) * 32 + 16 + n] = (_Float16)s1[r];
    }
    const v16h pa = load_a_frag(ps, 32);

    // ---- O += P @ V (B-frags from LDS Vtile) ----
#pragma unroll
    for (int t = 0; t < 4; ++t)
      o[t] = wmma_f16(pa, load_b_fragT(Vtile + t * 16 * 32, 32), o[t]);

    __syncthreads();   // protect K/V tiles before next iteration overwrites
  }

  // ---- normalize, scatter to attn16 [token][C], C-col = h*64 + d ----
  const int b = bh >> 4, h = bh & 15;
#pragma unroll
  for (int r = 0; r < 8; ++r) {
    const float inv = 1.0f / lrow[r];
    const int tok = b * SEQ + qt * 16 + r + 8 * hi;
#pragma unroll
    for (int t = 0; t < 4; ++t)
      attn16[(size_t)tok * DIM + h * HDIM + t * 16 + n] = (_Float16)(o[t][r] * inv);
  }
}

// ---- Kernel 5: out = attn16 @ w_proj + b_proj + x (fp32 out) ----------------
__global__ void k_proj_gemm(const _Float16* __restrict__ attn16,
                            const _Float16* __restrict__ wprojT,
                            const float* __restrict__ bias,
                            const float* __restrict__ x,
                            float* __restrict__ out) {
  const int wid  = blockIdx.x * (blockDim.x >> 5) + (threadIdx.x >> 5);
  const int row0 = (wid / 16) * 16;
  const int col0 = (wid % 16) * 64;
  v8f acc[4] = {v8f{}, v8f{}, v8f{}, v8f{}};
  for (int kk = 0; kk < DIM; kk += 32) {
    __builtin_prefetch(attn16 + (size_t)row0 * DIM + kk + 64, 0, 1);
    v16h a = load_a_frag(attn16 + (size_t)row0 * DIM + kk, DIM);
#pragma unroll
    for (int t = 0; t < 4; ++t) {
      v16h b = load_b_fragT(wprojT + (size_t)(col0 + t * 16) * DIM + kk, DIM);
      acc[t] = wmma_f16(a, b, acc[t]);
    }
  }
  const int lane = threadIdx.x & 31;
  const int n    = lane & 15;
  const int hi   = lane >> 4;
#pragma unroll
  for (int t = 0; t < 4; ++t) {
#pragma unroll
    for (int r = 0; r < 8; ++r) {
      const int col = col0 + t * 16 + n;
      const int row = row0 + r + 8 * hi;
      const size_t idx = (size_t)row * DIM + col;
      out[idx] = acc[t][r] + bias[col] + x[idx];
    }
  }
}

// ---- Host launcher ----------------------------------------------------------
extern "C" void kernel_launch(void* const* d_in, const int* in_sizes, int n_in,
                              void* d_out, int out_size, void* d_ws, size_t ws_size,
                              hipStream_t stream) {
  const float* x     = (const float*)d_in[0];   // [2,2048,1024]
  const float* wqkv  = (const float*)d_in[1];   // [1024,3072]
  const float* wproj = (const float*)d_in[2];   // [1024,1024]
  const float* bproj = (const float*)d_in[3];   // [1024]
  const float* gamma = (const float*)d_in[4];   // [1024]
  const float* beta  = (const float*)d_in[5];   // [1024]
  float* out = (float*)d_out;

  char* ws = (char*)d_ws;
  _Float16* h16    = (_Float16*)(ws);                       // 8 MiB
  _Float16* wqkvT  = (_Float16*)(ws + (8u << 20));          // 6 MiB
  _Float16* wprojT = (_Float16*)(ws + (14u << 20));         // 2 MiB
  _Float16* q      = (_Float16*)(ws + (16u << 20));         // 8 MiB
  _Float16* kmat   = (_Float16*)(ws + (24u << 20));         // 8 MiB
  _Float16* vt     = (_Float16*)(ws + (32u << 20));         // 8 MiB
  _Float16* attn16 = (_Float16*)(ws + (40u << 20));         // 8 MiB (48 total)

  k_ln_f16<<<TOK / 8, 256, 0, stream>>>(x, gamma, beta, h16);
  k_wtrans<<<(DIM * NCOL3) / 256, 256, 0, stream>>>(wqkv, wqkvT, DIM, NCOL3);
  k_wtrans<<<(DIM * DIM) / 256, 256, 0, stream>>>(wproj, wprojT, DIM, DIM);
  k_qkv_gemm<<<12288 / 8, 256, 0, stream>>>(h16, wqkvT, q, kmat, vt);
  k_attn<<<4096 / 4, 128, 0, stream>>>(q, kmat, vt, attn16);
  k_proj_gemm<<<4096 / 8, 256, 0, stream>>>(attn16, wprojT, bproj, x, out);
}
